// _CDAE_31155692765828
// MI455X (gfx1250) — matrix operand order repaired and checked by reading, not compiled, not run on hardware
//
#include <hip/hip_runtime.h>
#include <hip/hip_bf16.h>

typedef __attribute__((ext_vector_type(16))) _Float16 v16h;
typedef __attribute__((ext_vector_type(2)))  _Float16 h2;
typedef __attribute__((ext_vector_type(8)))  float    v8f;

#define DIMS 64

__device__ __forceinline__ unsigned rotl32(unsigned x, int n) {
  return (x << n) | (x >> (32 - n));
}

// JAX threefry2x32 (key injection schedule: ks0,ks1 ; ks1,ks2+1 ; ks2,ks0+2 ; ks0,ks1+3 ; ks1,ks2+4)
__device__ __forceinline__ void threefry2x32(unsigned k0, unsigned k1,
                                             unsigned x0, unsigned x1,
                                             unsigned& o0, unsigned& o1) {
  const unsigned ks0 = k0, ks1 = k1, ks2 = k0 ^ k1 ^ 0x1BD11BDAu;
  const int ra[4] = {13, 15, 26, 6};
  const int rb[4] = {17, 29, 16, 24};
  x0 += ks0; x1 += ks1;
#pragma unroll
  for (int i = 0; i < 4; ++i) { x0 += x1; x1 = rotl32(x1, ra[i]); x1 ^= x0; }
  x0 += ks1; x1 += ks2 + 1u;
#pragma unroll
  for (int i = 0; i < 4; ++i) { x0 += x1; x1 = rotl32(x1, rb[i]); x1 ^= x0; }
  x0 += ks2; x1 += ks0 + 2u;
#pragma unroll
  for (int i = 0; i < 4; ++i) { x0 += x1; x1 = rotl32(x1, ra[i]); x1 ^= x0; }
  x0 += ks0; x1 += ks1 + 3u;
#pragma unroll
  for (int i = 0; i < 4; ++i) { x0 += x1; x1 = rotl32(x1, rb[i]); x1 ^= x0; }
  x0 += ks1; x1 += ks2 + 4u;
  o0 = x0; o1 = x1;
}

// ---------------- stage 0: zero hidden accum, item mask, reg scalar ----------------
__global__ void k_zero(float* hidden, float* mask, float* reg, int nh, int nm) {
  int i = blockIdx.x * blockDim.x + threadIdx.x;
  if (i < nh) hidden[i] = 0.0f;
  if (i < nm) mask[i]   = 0.0f;
  if (i == 0) *reg = 0.0f;
}

// ---------------- stage 1: dropout values via threefry (matches jax.random.key(42)) ----------------
__global__ void k_vals(float* vals, int nnz) {
  int i = blockIdx.x * blockDim.x + threadIdx.x;
  if (i >= nnz) return;
  int half = nnz >> 1;
  bool second = (i >= half);
  unsigned lo = second ? (unsigned)(i - half) : (unsigned)i;
  unsigned hi = lo + (unsigned)half;
  unsigned c0, c1;
  threefry2x32(0u, 42u, lo, hi, c0, c1);
  unsigned bits = second ? c1 : c0;
  float u = __uint_as_float((bits >> 9) | 0x3f800000u) - 1.0f;   // [0,1)
  vals[i] = (u < 0.8f) ? 1.25f : 0.0f;                           // mask / keep
}

// ---------------- stage 2: sparse @ en_emb via scatter-add ----------------
__global__ void k_scatter(const int* __restrict__ sp_row, const int* __restrict__ sp_col,
                          const float* __restrict__ vals, const float* __restrict__ en_emb,
                          float* __restrict__ hidden, int nnz) {
  long long t = (long long)blockIdx.x * blockDim.x + threadIdx.x;
  if (t >= (long long)nnz * DIMS) return;
  int k = (int)(t >> 6), d = (int)(t & 63);
  float v = vals[k];
  if (v == 0.0f) return;
  int r = sp_row[k], c = sp_col[k];
  atomicAdd(&hidden[r * DIMS + d], v * en_emb[(long long)c * DIMS + d]);
}

// ---------------- stage 3: tanh(hidden + user_emb[uid] + offset) -> f16 ----------------
__global__ void k_activate(const float* __restrict__ hidden, const int* __restrict__ user_ids,
                           const float* __restrict__ user_emb, const float* __restrict__ en_offset,
                           _Float16* __restrict__ h16, int total) {
  int i = blockIdx.x * blockDim.x + threadIdx.x;
  if (i >= total) return;
  int b = i >> 6, d = i & 63;
  float t = tanhf(hidden[i] + user_emb[(long long)user_ids[b] * DIMS + d] + en_offset[d]);
  h16[i] = (_Float16)t;
}

// ---------------- stage 4: decode — 16 pairs/wave, diag(H x E^T) via WMMA f16 ----------------
__global__ __launch_bounds__(256) void k_decode(const _Float16* __restrict__ h16,
                                                const float* __restrict__ de_emb,
                                                const float* __restrict__ de_bias,
                                                const int* __restrict__ bat_idx,
                                                const int* __restrict__ bat_items,
                                                float* __restrict__ ratings, int n) {
  __shared__ __align__(16) _Float16 sh[8][16 * DIMS];  // 16 hidden rows per wave (f16)
  __shared__ __align__(16) _Float16 se[8][16 * DIMS];  // 16 item rows per wave (f16)
  int w = threadIdx.x >> 5;
  int lane = threadIdx.x & 31;
  int t0 = (blockIdx.x * 8 + w) * 16;

  // ---- stage 16 h-rows and 16 e-rows into LDS (coalesced 128B h-row, 256B e-row loads) ----
  for (int m = 0; m < 16; ++m) {
    int t = t0 + m; if (t >= n) t = n - 1;
    long long brow = bat_idx[t];
    ((unsigned*)&sh[w][m * DIMS])[lane] =
        ((const unsigned*)(h16 + brow * DIMS))[lane];
    long long item = bat_items[t];
    float2 e = ((const float2*)(de_emb + item * DIMS))[lane];
    h2 p; p.x = (_Float16)e.x; p.y = (_Float16)e.y;
    ((h2*)&se[w][m * DIMS])[lane] = p;
  }
  __syncthreads();

  // ---- build fragments per documented 16-bit layouts ----
  // A (16x32, f16): lane holds row M=lane&15; lanes<16 carry K=kb+{0..7,16..23}, kb=0; lanes>=16 kb=8.
  int r  = lane & 15;
  int kb = (lane < 16) ? 0 : 8;
  const _Float16* hrow = &sh[w][r * DIMS];
  v16h a0, a1, b0v, b1v;
#pragma unroll
  for (int i = 0; i < 8; ++i) {
    a0[i]     = hrow[kb + i];
    a0[i + 8] = hrow[kb + 16 + i];
    a1[i]     = hrow[32 + kb + i];
    a1[i + 8] = hrow[32 + kb + 16 + i];
  }
  // B (32x16, f16): lane holds column N=lane&15, same K order as A; B[k][nn] = E[nn][k0+k].
  const _Float16* erow = &se[w][r * DIMS];
#pragma unroll
  for (int i = 0; i < 8; ++i) {
    b0v[i]     = erow[kb + i];
    b0v[i + 8] = erow[kb + 16 + i];
    b1v[i]     = erow[32 + kb + i];
    b1v[i + 8] = erow[32 + kb + 16 + i];
  }

  v8f c = {};
  c = __builtin_amdgcn_wmma_f32_16x16x32_f16(false, a0, false, b0v, (short)0, c, false, false);
  c = __builtin_amdgcn_wmma_f32_16x16x32_f16(false, a1, false, b1v, (short)0, c, false, false);

  // ---- extract diagonal: D[m][n], lanes 0-15: n=lane, m=vgpr; lanes 16-31: n=lane-16, m=vgpr+8 ----
  int j = lane & 7;
  float rd = c[0];
  if (j == 1) rd = c[1];
  if (j == 2) rd = c[2];
  if (j == 3) rd = c[3];
  if (j == 4) rd = c[4];
  if (j == 5) rd = c[5];
  if (j == 6) rd = c[6];
  if (j == 7) rd = c[7];

  bool wr = false; int nidx = 0;
  if (lane < 8)        { wr = true; nidx = lane; }        // diag n=0..7
  else if (lane >= 24) { wr = true; nidx = lane - 16; }   // diag n=8..15
  if (wr) {
    int t = t0 + nidx;
    if (t < n) ratings[t] = rd + de_bias[bat_items[t]];
  }
}

// ---------------- stage 5: item mask scatter ----------------
__global__ void k_mask(const int* __restrict__ bat_items, float* __restrict__ mask, int n) {
  int i = blockIdx.x * blockDim.x + threadIdx.x;
  if (i < n) mask[bat_items[i]] = 1.0f;
}

// ---------------- stage 6: reg loss reduction ----------------
__global__ __launch_bounds__(256) void k_reg(const float* __restrict__ mask,
                                             const float* __restrict__ en_emb,
                                             const float* __restrict__ de_emb,
                                             const float* __restrict__ de_bias,
                                             const int* __restrict__ user_ids,
                                             const float* __restrict__ user_emb,
                                             const float* __restrict__ en_offset,
                                             float* __restrict__ reg, int num_items, int bsz) {
  __shared__ float red[256];
  int i = blockIdx.x * blockDim.x + threadIdx.x;
  float s = 0.0f;
  if (i < num_items) {
    if (mask[i] != 0.0f) {
      const float4* e4 = (const float4*)(en_emb + (long long)i * DIMS);
      const float4* d4 = (const float4*)(de_emb + (long long)i * DIMS);
#pragma unroll
      for (int q = 0; q < 16; ++q) {
        float4 a = e4[q], b = d4[q];
        s += a.x * a.x + a.y * a.y + a.z * a.z + a.w * a.w;
        s += b.x * b.x + b.y * b.y + b.z * b.z + b.w * b.w;
      }
      float bb = de_bias[i];
      s += bb * bb;
    }
  } else if (i < num_items + bsz) {
    int b = i - num_items;
    const float4* u4 = (const float4*)(user_emb + (long long)user_ids[b] * DIMS);
#pragma unroll
    for (int q = 0; q < 16; ++q) {
      float4 a = u4[q];
      s += a.x * a.x + a.y * a.y + a.z * a.z + a.w * a.w;
    }
  } else if (i == num_items + bsz) {
    for (int q = 0; q < DIMS; ++q) { float v = en_offset[q]; s += v * v; }
  }
  s *= 0.5f;
  red[threadIdx.x] = s;
  __syncthreads();
  for (int st = 128; st > 0; st >>= 1) {
    if (threadIdx.x < st) red[threadIdx.x] += red[threadIdx.x + st];
    __syncthreads();
  }
  if (threadIdx.x == 0 && red[0] != 0.0f) atomicAdd(reg, red[0]);
}

extern "C" void kernel_launch(void* const* d_in, const int* in_sizes, int n_in,
                              void* d_out, int out_size, void* d_ws, size_t ws_size,
                              hipStream_t stream) {
  const int*   user_ids  = (const int*)d_in[0];
  const int*   bat_idx   = (const int*)d_in[1];
  const int*   sp_row    = (const int*)d_in[2];
  const int*   sp_col    = (const int*)d_in[3];
  const int*   bat_items = (const int*)d_in[4];
  const float* en_emb    = (const float*)d_in[5];
  const float* en_offset = (const float*)d_in[6];
  const float* de_emb    = (const float*)d_in[7];
  const float* de_bias   = (const float*)d_in[8];
  const float* user_emb  = (const float*)d_in[9];

  const int Bsz  = in_sizes[0];
  const int N    = in_sizes[1];
  const int NNZv = in_sizes[2];
  const int NUMI = in_sizes[8];
  const int HID  = Bsz * DIMS;

  float* out     = (float*)d_out;
  float* ratings = out;
  float* reg     = out + (out_size - 1);

  // workspace carve-out (~2.8 MB)
  char* ws = (char*)d_ws;
  size_t off = 0;
  auto al = [](size_t x) { return (x + 255) & ~(size_t)255; };
  float*    hidden = (float*)(ws + off);    off = al(off + (size_t)HID * 4);
  float*    vals   = (float*)(ws + off);    off = al(off + (size_t)NNZv * 4);
  float*    maskp  = (float*)(ws + off);    off = al(off + (size_t)NUMI * 4);
  _Float16* h16    = (_Float16*)(ws + off); off = al(off + (size_t)HID * 2);

  int zn = (HID > NUMI) ? HID : NUMI;
  k_zero<<<(zn + 255) / 256, 256, 0, stream>>>(hidden, maskp, reg, HID, NUMI);

  k_vals<<<(NNZv + 255) / 256, 256, 0, stream>>>(vals, NNZv);

  long long selems = (long long)NNZv * DIMS;
  k_scatter<<<(int)((selems + 255) / 256), 256, 0, stream>>>(sp_row, sp_col, vals, en_emb, hidden, NNZv);

  k_activate<<<(HID + 255) / 256, 256, 0, stream>>>(hidden, user_ids, user_emb, en_offset, h16, HID);

  int tiles = (N + 127) / 128;  // 128 ratings per 256-thread block (8 waves x 16)
  k_decode<<<tiles, 256, 0, stream>>>(h16, de_emb, de_bias, bat_idx, bat_items, ratings, N);

  k_mask<<<(N + 255) / 256, 256, 0, stream>>>(bat_items, maskp, N);

  int rt = NUMI + Bsz + 1;
  k_reg<<<(rt + 255) / 256, 256, 0, stream>>>(maskp, en_emb, de_emb, de_bias,
                                              user_ids, user_emb, en_offset, reg, NUMI, Bsz);
}
